// GCT_23021024706960
// MI455X (gfx1250) — matrix operand order repaired and checked by reading, not compiled, not run on hardware
//
#include <hip/hip_runtime.h>
#include <hip/hip_bf16.h>

typedef __attribute__((ext_vector_type(16))) __bf16 v16bf;
typedef __attribute__((ext_vector_type(8)))  __bf16 v8bf;
typedef __attribute__((ext_vector_type(8)))  float  v8f;

#define B_    4
#define N_    2048
#define KNB   32
#define NK    (N_ * KNB)      // 65536
#define CF    64
#define CIN   6
#define CO    64
#define NP    4               // points per workgroup
#define NCOL  (NP * KNB)      // 128 columns per workgroup

// ---------------------------------------------------------------- setup kernels

// BN folding: scale = g / sqrt(v + eps); shift = b - m*scale.
// permute!=0 remaps channel r = co*6+ci  ->  idx = ci*64+co (for Wattn epilogue fusion).
__global__ void bn_prep(const float* __restrict__ g, const float* __restrict__ bb,
                        const float* __restrict__ m, const float* __restrict__ v,
                        float* __restrict__ s, float* __restrict__ t, int n, int permute) {
    int i = blockIdx.x * blockDim.x + threadIdx.x;
    if (i >= n) return;
    int src = i;
    if (permute) { int ci = i >> 6; int co = i & 63; src = co * 6 + ci; }
    float sc = g[src] * rsqrtf(v[src] + 1e-5f);
    s[i] = sc;
    t[i] = bb[src] - m[src] * sc;
}

// x_q = lrelu(bn(Wq @ feat)) : [B,64,N]
__global__ void xq_kernel(const float* __restrict__ Wq, const float* __restrict__ feat,
                          const float* __restrict__ s, const float* __restrict__ t,
                          float* __restrict__ xq) {
    int i = blockIdx.x * blockDim.x + threadIdx.x;
    if (i >= B_ * 64 * N_) return;
    int n = i % N_;
    int c = (i / N_) & 63;
    int b = i / (N_ * 64);
    float acc = 0.f;
#pragma unroll
    for (int j = 0; j < 32; ++j)
        acc += Wq[c * 32 + j] * feat[(b * 32 + j) * N_ + n];
    float vv = acc * s[c] + t[c];
    xq[i] = vv > 0.f ? vv : 0.2f * vv;
}

// pos3 = lrelu(bn(Wpos1 @ points)) : [B,6,N]
__global__ void pos3_kernel(const float* __restrict__ Wpos1, const float* __restrict__ pts,
                            const float* __restrict__ s, const float* __restrict__ t,
                            float* __restrict__ pos3) {
    int i = blockIdx.x * blockDim.x + threadIdx.x;
    if (i >= B_ * CIN * N_) return;
    int n = i % N_;
    int c = (i / N_) % CIN;
    int b = i / (N_ * CIN);
    float acc = 0.f;
#pragma unroll
    for (int j = 0; j < 3; ++j)
        acc += Wpos1[c * 3 + j] * pts[(b * 3 + j) * N_ + n];
    float vv = acc * s[c] + t[c];
    pos3[i] = vv > 0.f ? vv : 0.2f * vv;
}

// weights -> bf16 with BN scale folded into rows.
// Wattn permuted to 6 blocks of [64co x 64cf], row (ci,co) scaled by bna_s[ci*64+co].
// Wlin rows scaled by bnl_s[o]; Wres rows scaled by bnr_s[o].
__global__ void wcvt_kernel(const float* __restrict__ Wattn, const float* __restrict__ Wlin,
                            const float* __restrict__ Wres,
                            const float* __restrict__ bna_s, const float* __restrict__ bnl_s,
                            const float* __restrict__ bnr_s,
                            __bf16* __restrict__ wattnp, __bf16* __restrict__ wlin,
                            __bf16* __restrict__ wres) {
    int i = blockIdx.x * blockDim.x + threadIdx.x;
    if (i < 6 * 64 * 64) {
        int ci = i >> 12;
        int r  = i & 4095;
        int co = r >> 6, cf = r & 63;
        wattnp[i] = (__bf16)(Wattn[(co * 6 + ci) * 64 + cf] * bna_s[ci * 64 + co]);
    } else if (i < 6 * 4096 + 4096) {
        int j = i - 6 * 4096;
        wlin[j] = (__bf16)(Wlin[j] * bnl_s[j >> 6]);
    } else if (i < 6 * 4096 + 2 * 4096) {
        int j = i - 7 * 4096;
        wres[j] = (__bf16)(Wres[j] * bnr_s[j >> 6]);
    }
}

// ---------------------------------------------------------------- fragment loads

// A fragment (16x32 bf16, row-major source, 64-element row stride):
// lane l holds row (l&15); k = 16*(l>>4) .. +15 within kt*32.
__device__ __forceinline__ v16bf loadA(const __bf16* __restrict__ w, int m, int kt, int lane) {
    int off = (m * 16 + (lane & 15)) * 64 + kt * 32 + 16 * (lane >> 4);
    v8bf c0 = *(const v8bf*)(w + off);
    v8bf c1 = *(const v8bf*)(w + off + 8);
    v16bf a;
#pragma unroll
    for (int i = 0; i < 8; ++i) { a[i] = c0[i]; a[i + 8] = c1[i]; }
    return a;
}

// B fragment (32x16 bf16) from col-major LDS (64 channels contiguous per column):
// lane l holds col (l&15), k = kt*32 + 16*(l>>4) .. +15  (one contiguous 32B read).
__device__ __forceinline__ v16bf loadB(const __bf16* buf, int c0, int kt, int lane) {
    return *(const v16bf*)(buf + (c0 + (lane & 15)) * 64 + kt * 32 + 16 * (lane >> 4));
}

__device__ __forceinline__ v8f wmma_bf16(v16bf a, v16bf b, v8f c) {
    return __builtin_amdgcn_wmma_f32_16x16x32_bf16(false, a, false, b, (short)0, c, false, false);
}

// ---------------------------------------------------------------- fused main kernel

__global__ __launch_bounds__(256)
void fused_pt_attn(const float* __restrict__ x, const float* __restrict__ y,
                   const float* __restrict__ Wpos2,
                   const float* __restrict__ xq, const float* __restrict__ pos3,
                   const __bf16* __restrict__ wattnp, const __bf16* __restrict__ wlin,
                   const __bf16* __restrict__ wres,
                   const float* __restrict__ bna_t,
                   const float* __restrict__ bn1_s, const float* __restrict__ bn1_t,
                   const float* __restrict__ bnl_t, const float* __restrict__ bnr_t,
                   float* __restrict__ out) {
    __shared__ __attribute__((aligned(32))) __bf16 ybuf[NCOL * 64];  // y tile, col-major
    __shared__ __attribute__((aligned(32))) __bf16 abuf[NCOL * 64];  // attn, later h
    __shared__ float xcol[CIN][NCOL];
    __shared__ float xq_s[64][NP];
    __shared__ float pos3_s[CIN][NP];
    __shared__ float wpos2_s[64][CIN];
    __shared__ float tA[6 * 64];
    __shared__ float s1[64], t1[64], tlr[64];   // tlr = bnl_t + bnr_t (pre-combined)

    const int tid  = threadIdx.x;
    const int lane = tid & 31;
    const int wave = tid >> 5;
    const int b    = blockIdx.x >> 9;            // / (N/NP)
    const int nb   = blockIdx.x & 511;
    const int n0   = nb * NP;
    const int g0   = n0 * KNB;                   // column offset in NK plane

    // warm L2 with the (shared) bf16 weight block: 66KB -> global_prefetch_b8
    if (blockIdx.x < 8) {
        const char* wp = (const char*)wattnp;
        __builtin_prefetch(wp + tid * 128, 0, 1);
        __builtin_prefetch(wp + 32768 + tid * 128, 0, 1);
    }

    // ---- stage 0: constants + per-point precomputed tensors into LDS
    for (int i = tid; i < 384; i += 256) {
        tA[i] = bna_t[i];
        wpos2_s[i / 6][i % 6] = Wpos2[i];
    }
    for (int i = tid; i < 64; i += 256) {
        s1[i] = bn1_s[i]; t1[i] = bn1_t[i];
        tlr[i] = bnl_t[i] + bnr_t[i];
    }
    {
        int c = tid >> 2, p = tid & 3;                       // 64*NP == 256
        xq_s[c][p] = xq[(b * 64 + c) * N_ + n0 + p];
        if (tid < CIN * NP) pos3_s[tid >> 2][tid & 3] = pos3[(b * CIN + (tid >> 2)) * N_ + n0 + (tid & 3)];
    }
    for (int i = tid; i < CIN * NCOL; i += 256) {
        int ci = i / NCOL, col = i % NCOL;
        xcol[ci][col] = x[(b * CIN + ci) * NK + g0 + col];
    }
    // ---- stage 1: y tile -> LDS bf16 col-major (coalesced global reads)
    for (int i = tid; i < 64 * NCOL; i += 256) {
        int cf = i >> 7, col = i & 127;
        ybuf[col * 64 + cf] = (__bf16)y[(b * 64 + cf) * NK + g0 + col];
    }
    __syncthreads();

    // ---- stage 2: attn_pre[cf,col] = xq - y + Wpos2 @ (pos3 - x)
    {
        int col = tid & 127;
        int nl  = col >> 5;
        float dp[CIN];
#pragma unroll
        for (int ci = 0; ci < CIN; ++ci) dp[ci] = pos3_s[ci][nl] - xcol[ci][col];
        int cf0 = (tid >> 7) * 32;
        for (int cf = cf0; cf < cf0 + 32; ++cf) {
            float v = xq_s[cf][nl] - (float)ybuf[col * 64 + cf];
#pragma unroll
            for (int ci = 0; ci < CIN; ++ci) v += wpos2_s[cf][ci] * dp[ci];
            abuf[col * 64 + cf] = (__bf16)v;
        }
    }
    __syncthreads();

    // ---- stage 3: softmax over K=32 neighbors (scale 1/sqrt(64) = 0.125)
    {
        int nl = tid >> 6;          // NP*64 == 256 items, one per thread
        int cf = tid & 63;
        float vals[KNB];
        float mx = -3.0e38f;
#pragma unroll
        for (int k = 0; k < KNB; ++k) {
            vals[k] = (float)abuf[(nl * 32 + k) * 64 + cf];
            mx = vals[k] > mx ? vals[k] : mx;
        }
        float sum = 0.f;
#pragma unroll
        for (int k = 0; k < KNB; ++k) {
            vals[k] = __expf((vals[k] - mx) * 0.125f);
            sum += vals[k];
        }
        float inv = 1.f / sum;
#pragma unroll
        for (int k = 0; k < KNB; ++k)
            abuf[(nl * 32 + k) * 64 + cf] = (__bf16)(vals[k] * inv);
    }
    __syncthreads();

    // ---- stage 4: out[co,col] = sum_ci ((sW_ci @ softattn)[co,col] + tA) * x[ci,col]
    // (BN scale pre-folded into weights.)  32 (m,c) tiles; 4 per wave.
    float accv[4][8];
    const int hi = lane >> 4;
#pragma unroll
    for (int i = 0; i < 4; ++i) {
        int p = wave * 4 + i;
        int m = p >> 3, c = p & 7;
        int colL = c * 16 + (lane & 15);
        v16bf bA0 = loadB(abuf, c * 16, 0, lane);
        v16bf bA1 = loadB(abuf, c * 16, 1, lane);
        float acc[8];
#pragma unroll
        for (int vv = 0; vv < 8; ++vv) acc[vv] = 0.f;
#pragma unroll
        for (int ci = 0; ci < CIN; ++ci) {
            const __bf16* w = wattnp + ci * 4096;
            v16bf a0 = loadA(w, m, 0, lane);
            v16bf a1 = loadA(w, m, 1, lane);
            v8f d;
#pragma unroll
            for (int vv = 0; vv < 8; ++vv) d[vv] = 0.f;
            d = wmma_bf16(a0, bA0, d);
            d = wmma_bf16(a1, bA1, d);
            float xf = xcol[ci][colL];
#pragma unroll
            for (int vv = 0; vv < 8; ++vv) {
                int row = m * 16 + vv + 8 * hi;
                acc[vv] += (d[vv] + tA[ci * 64 + row]) * xf;
            }
        }
#pragma unroll
        for (int vv = 0; vv < 8; ++vv) accv[i][vv] = acc[vv];
    }
    __syncthreads();   // all reads of abuf (softattn) done

    // ---- stage 4b: h = lrelu(bn1(out)) -> abuf (reused as h buffer, bf16 col-major)
#pragma unroll
    for (int i = 0; i < 4; ++i) {
        int p = wave * 4 + i;
        int m = p >> 3, c = p & 7;
        int colL = c * 16 + (lane & 15);
#pragma unroll
        for (int vv = 0; vv < 8; ++vv) {
            int row = m * 16 + vv + 8 * hi;
            float h = accv[i][vv] * s1[row] + t1[row];
            h = h > 0.f ? h : 0.2f * h;
            abuf[colL * 64 + row] = (__bf16)h;
        }
    }
    __syncthreads();

    // ---- stage 5: final = lrelu((sWlin @ h) + (sWres @ y) + (tl + tr))
#pragma unroll
    for (int i = 0; i < 4; ++i) {
        int p = wave * 4 + i;
        int m = p >> 3, c = p & 7;
        int colL = c * 16 + (lane & 15);
        v16bf h0 = loadB(abuf, c * 16, 0, lane);
        v16bf h1 = loadB(abuf, c * 16, 1, lane);
        v16bf y0 = loadB(ybuf, c * 16, 0, lane);
        v16bf y1 = loadB(ybuf, c * 16, 1, lane);
        v16bf la0 = loadA(wlin, m, 0, lane);
        v16bf la1 = loadA(wlin, m, 1, lane);
        v16bf ra0 = loadA(wres, m, 0, lane);
        v16bf ra1 = loadA(wres, m, 1, lane);
        v8f d;
#pragma unroll
        for (int vv = 0; vv < 8; ++vv) d[vv] = 0.f;
        d = wmma_bf16(la0, h0, d);
        d = wmma_bf16(la1, h1, d);
        d = wmma_bf16(ra0, y0, d);
        d = wmma_bf16(ra1, y1, d);
#pragma unroll
        for (int vv = 0; vv < 8; ++vv) {
            int row = m * 16 + vv + 8 * hi;
            float f = d[vv] + tlr[row];
            f = f > 0.f ? f : 0.2f * f;
            out[(b * 64 + row) * NK + g0 + colL] = f;
        }
    }
}

// ---------------------------------------------------------------- launch

extern "C" void kernel_launch(void* const* d_in, const int* in_sizes, int n_in,
                              void* d_out, int out_size, void* d_ws, size_t ws_size,
                              hipStream_t stream) {
    (void)in_sizes; (void)n_in; (void)out_size; (void)ws_size;

    const float* x      = (const float*)d_in[0];
    const float* y      = (const float*)d_in[1];
    const float* points = (const float*)d_in[2];
    const float* feat   = (const float*)d_in[3];
    const float* Wq     = (const float*)d_in[4];
    const float* Wpos1  = (const float*)d_in[5];
    const float* Wpos2  = (const float*)d_in[6];
    const float* Wattn  = (const float*)d_in[7];
    const float* Wlin   = (const float*)d_in[8];
    const float* Wres   = (const float*)d_in[9];
    // BN params: g,b,m,v for q(10), p(14), a(18), 1(22), l(26), r(30)

    float* w = (float*)d_ws;
    float* xq     = w;              w += B_ * 64 * N_;     // 524288
    float* pos3   = w;              w += B_ * CIN * N_;    // 49152
    float* bnq_s  = w; w += 64;  float* bnq_t = w; w += 64;
    float* bnp_s  = w; w += 8;   float* bnp_t = w; w += 8;
    float* bna_s  = w; w += 384; float* bna_t = w; w += 384;
    float* bn1_s  = w; w += 64;  float* bn1_t = w; w += 64;
    float* bnl_s  = w; w += 64;  float* bnl_t = w; w += 64;
    float* bnr_s  = w; w += 64;  float* bnr_t = w; w += 64;
    __bf16* wattnp = (__bf16*)w;
    __bf16* wlin   = wattnp + 6 * 4096;
    __bf16* wres   = wlin + 4096;

    // BN folding
    bn_prep<<<1, 64, 0, stream>>>((const float*)d_in[10], (const float*)d_in[11],
                                  (const float*)d_in[12], (const float*)d_in[13],
                                  bnq_s, bnq_t, 64, 0);
    bn_prep<<<1, 64, 0, stream>>>((const float*)d_in[14], (const float*)d_in[15],
                                  (const float*)d_in[16], (const float*)d_in[17],
                                  bnp_s, bnp_t, 6, 0);
    bn_prep<<<6, 64, 0, stream>>>((const float*)d_in[18], (const float*)d_in[19],
                                  (const float*)d_in[20], (const float*)d_in[21],
                                  bna_s, bna_t, 384, 1);
    bn_prep<<<1, 64, 0, stream>>>((const float*)d_in[22], (const float*)d_in[23],
                                  (const float*)d_in[24], (const float*)d_in[25],
                                  bn1_s, bn1_t, 64, 0);
    bn_prep<<<1, 64, 0, stream>>>((const float*)d_in[26], (const float*)d_in[27],
                                  (const float*)d_in[28], (const float*)d_in[29],
                                  bnl_s, bnl_t, 64, 0);
    bn_prep<<<1, 64, 0, stream>>>((const float*)d_in[30], (const float*)d_in[31],
                                  (const float*)d_in[32], (const float*)d_in[33],
                                  bnr_s, bnr_t, 64, 0);

    // weight conversion (permuted Wattn, BN scale folded) and per-point precompute
    wcvt_kernel<<<128, 256, 0, stream>>>(Wattn, Wlin, Wres, bna_s, bnl_s, bnr_s,
                                         wattnp, wlin, wres);
    xq_kernel<<<(B_ * 64 * N_) / 256, 256, 0, stream>>>(Wq, feat, bnq_s, bnq_t, xq);
    pos3_kernel<<<(B_ * CIN * N_) / 256, 256, 0, stream>>>(Wpos1, points, bnp_s, bnp_t, pos3);

    // fused attention kernel: one workgroup per (b, 4-point block)
    fused_pt_attn<<<B_ * (N_ / NP), 256, 0, stream>>>(
        x, y, Wpos2, xq, pos3, wattnp, wlin, wres,
        bna_t, bn1_s, bn1_t, bnl_t, bnr_t,
        (float*)d_out);
}
